// MultiHeadAttention_32066225832159
// MI455X (gfx1250) — matrix operand order repaired and checked by reading, not compiled, not run on hardware
//
#include <hip/hip_runtime.h>
#include <hip/hip_bf16.h>

typedef __bf16 bf16;
typedef __attribute__((ext_vector_type(16))) __bf16 v16bf;
typedef __attribute__((ext_vector_type(8)))  __bf16 v8bf;
typedef __attribute__((ext_vector_type(4)))  __bf16 v4bf;
typedef __attribute__((ext_vector_type(8)))  float  v8f;

static __device__ __forceinline__ v8f zero_v8f() {
    v8f v;
#pragma unroll
    for (int i = 0; i < 8; ++i) v[i] = 0.0f;
    return v;
}

// Build a 16-element bf16 fragment from two contiguous 8-element (16B) LDS chunks.
static __device__ __forceinline__ v16bf load_frag2(const bf16* p0, const bf16* p1) {
    v8bf lo = *(const v8bf*)p0;
    v8bf hi = *(const v8bf*)p1;
    return __builtin_shufflevector(lo, hi, 0,1,2,3,4,5,6,7,8,9,10,11,12,13,14,15);
}

static __device__ __forceinline__ v8f wmma_bf16(v16bf a, v16bf b, v8f c) {
    return __builtin_amdgcn_wmma_f32_16x16x32_bf16(
        /*neg_a=*/false, a, /*neg_b=*/false, b,
        /*c_mod=*/(short)0, c, /*reuse_a=*/false, /*reuse_b=*/false);
}

// --- CDNA5 async global->LDS copy (GVS mode), tracked by ASYNCcnt ----------
// Per lane: LDS[lds_addr] = MEM[base + gofs], 16 bytes.
static __device__ __forceinline__ void async_ld16(unsigned lds_addr, unsigned gofs,
                                                  const void* base) {
    asm volatile("global_load_async_to_lds_b128 %0, %1, %2"
                 :: "v"(lds_addr), "v"(gofs), "s"(base) : "memory");
}
static __device__ __forceinline__ void wait_async0() {
    asm volatile("s_wait_asynccnt 0x0" ::: "memory");
}
// Low 32 bits of a generic LDS pointer == LDS byte offset (aperture encoding).
static __device__ __forceinline__ unsigned lds_off(const void* p) {
    return (unsigned)(unsigned long long)p;
}

// ---------------------------------------------------------------------------
// Stage 0a: f32 -> bf16 cast (vectorized: float4 in, 4x bf16 out)
// ---------------------------------------------------------------------------
__global__ __launch_bounds__(256)
void cast_f32_bf16_kernel(const float* __restrict__ in, bf16* __restrict__ out, int n4) {
    int i = blockIdx.x * 256 + threadIdx.x;
    if (i >= n4) return;
    float4 v = ((const float4*)in)[i];
    v4bf o = {(bf16)v.x, (bf16)v.y, (bf16)v.z, (bf16)v.w};
    ((v4bf*)out)[i] = o;
}

// ---------------------------------------------------------------------------
// Stage 0b: W[K][N] f32 -> Wt[N][K] bf16 (so GEMM B-fragments are contiguous)
// ---------------------------------------------------------------------------
__global__ __launch_bounds__(256)
void transpose_cast_kernel(const float* __restrict__ W, bf16* __restrict__ Wt,
                           int K, int N) {
    long long idx = (long long)blockIdx.x * 256 + threadIdx.x;
    if (idx >= (long long)K * N) return;
    int k = (int)(idx / N);
    int n = (int)(idx - (long long)k * N);
    Wt[(size_t)n * K + k] = (bf16)W[idx];
}

// ---------------------------------------------------------------------------
// Stage 1 & 4: bf16 GEMM  C[M,N] = A[M,K] * Bt[N,K]^T + bias
// Block tile 128x128, K-step 64, 256 threads = 8 waves (2x4), 16 WMMA / step / wave
// Tiles staged with GLOBAL_LOAD_ASYNC_TO_LDS_B128.
// ---------------------------------------------------------------------------
template<bool OUT_BF16>
__global__ __launch_bounds__(256)
void gemm_kernel(const bf16* __restrict__ A, const bf16* __restrict__ Bt,
                 const float* __restrict__ bias, void* __restrict__ Cout,
                 int M, int N, int K) {
    __shared__ bf16 As[128 * 64];
    __shared__ bf16 Bs[128 * 64];

    const int tid  = threadIdx.x;
    const int lane = tid & 31;
    const int w    = tid >> 5;
    const int mW   = w & 1;          // wave's M position (2)
    const int nW   = w >> 1;         // wave's N position (4)
    const int hl   = lane >> 4;      // half-wave
    const int ln   = lane & 15;

    const int m0 = blockIdx.y * 128;
    const int n0 = blockIdx.x * 128;

    const unsigned ldsA0 = lds_off(As);
    const unsigned ldsB0 = lds_off(Bs);

    v8f acc[4][2];
#pragma unroll
    for (int a = 0; a < 4; ++a)
#pragma unroll
        for (int b = 0; b < 2; ++b) acc[a][b] = zero_v8f();

    for (int k0 = 0; k0 < K; k0 += 64) {
        // Async-stage tiles: 128 rows x 128B each for A and Bt (16B per lane-op).
#pragma unroll
        for (int i = tid; i < 1024; i += 256) {
            const int r = i >> 3, s = i & 7;
            const unsigned ga = (unsigned)(((size_t)(m0 + r) * K + k0 + s * 8) * sizeof(bf16));
            const unsigned gb = (unsigned)(((size_t)(n0 + r) * K + k0 + s * 8) * sizeof(bf16));
            async_ld16(ldsA0 + (unsigned)i * 16u, ga, (const void*)A);
            async_ld16(ldsB0 + (unsigned)i * 16u, gb, (const void*)Bt);
        }
        // Prefetch next K-step tiles into L2 while this one is consumed.
        if (k0 + 64 < K) {
            __builtin_prefetch(A  + (size_t)(m0 + (tid >> 1)) * K + k0 + 64, 0, 1);
            __builtin_prefetch(Bt + (size_t)(n0 + (tid >> 1)) * K + k0 + 64, 0, 1);
        }
        wait_async0();
        __syncthreads();

#pragma unroll
        for (int kc = 0; kc < 64; kc += 32) {
            // A fragments: lane holds row m=ln, K = {8*hl..8*hl+7, 16+8*hl..}
            v16bf af[4];
#pragma unroll
            for (int mi = 0; mi < 4; ++mi) {
                const bf16* ar = &As[(mW * 64 + mi * 16 + ln) * 64 + kc + hl * 8];
                af[mi] = load_frag2(ar, ar + 16);
            }
            // B fragments: lane holds column n=ln, K = {8*hl..8*hl+7, 16+8*hl..}
#pragma unroll
            for (int ni = 0; ni < 2; ++ni) {
                const bf16* br = &Bs[(nW * 32 + ni * 16 + ln) * 64 + kc + hl * 8];
                v16bf bfrag = load_frag2(br, br + 16);
#pragma unroll
                for (int mi = 0; mi < 4; ++mi)
                    acc[mi][ni] = wmma_bf16(af[mi], bfrag, acc[mi][ni]);
            }
        }
        __syncthreads();
    }

    // Epilogue: C layout -> lane(ln)=col, vgpr v -> row (v + 8*hl) of each 16x16 tile
#pragma unroll
    for (int mi = 0; mi < 4; ++mi) {
#pragma unroll
        for (int ni = 0; ni < 2; ++ni) {
            const int mg0 = m0 + mW * 64 + mi * 16 + hl * 8;
            const int ng  = n0 + nW * 32 + ni * 16 + ln;
            const float bv = bias[ng];
#pragma unroll
            for (int v = 0; v < 8; ++v) {
                float val = acc[mi][ni][v] + bv;
                if (OUT_BF16)
                    ((bf16*)Cout)[(size_t)(mg0 + v) * N + ng] = (bf16)val;
                else
                    ((float*)Cout)[(size_t)(mg0 + v) * N + ng] = val;
            }
        }
    }
}

// ---------------------------------------------------------------------------
// Stage 2: RoPE on q/k + split qkv [B*T][3*D] -> Q/K/V [B*H][T][DK] bf16
// One thread per (b,h,t, dk-pair)
// ---------------------------------------------------------------------------
__global__ __launch_bounds__(256)
void rope_split_kernel(const bf16* __restrict__ qkv,
                       bf16* __restrict__ Q, bf16* __restrict__ Ko,
                       bf16* __restrict__ Vo) {
    const int idx = blockIdx.x * 256 + threadIdx.x;   // < 2*16*2048*32 = 2^21
    const int i  = idx & 31;            // dk pair
    const int t  = (idx >> 5) & 2047;
    const int bh = idx >> 16;           // 0..31
    const int b  = bh >> 4;
    const int h  = bh & 15;

    const size_t row = (size_t)(b * 2048 + t) * 3072;
    const int cq = h * 64 + 2 * i;
    float q0 = (float)qkv[row + cq],        q1 = (float)qkv[row + cq + 1];
    float k0 = (float)qkv[row + 1024 + cq], k1 = (float)qkv[row + 1024 + cq + 1];
    float v0 = (float)qkv[row + 2048 + cq], v1 = (float)qkv[row + 2048 + cq + 1];

    // inv_freq = 10000^(-2i/64)
    const float inv = __expf(-((float)(2 * i) * (1.0f / 64.0f)) * 9.2103403719761836f);
    const float ang = (float)t * inv;
    float s, c;
    __sincosf(ang, &s, &c);

    const size_t orow = ((size_t)bh * 2048 + t) * 64 + 2 * i;
    Q[orow]      = (bf16)(q0 * c - q1 * s);
    Q[orow + 1]  = (bf16)(q0 * s + q1 * c);
    Ko[orow]     = (bf16)(k0 * c - k1 * s);
    Ko[orow + 1] = (bf16)(k0 * s + k1 * c);
    Vo[orow]     = (bf16)v0;
    Vo[orow + 1] = (bf16)v1;
}

// ---------------------------------------------------------------------------
// Stage 3: causal flash attention. Grid: (T/64 q-blocks, B*H).
// 128 threads = 4 waves; wave w owns q rows [qb*64 + 16w, +15].
// S-blocks of 32 keys; scores 16x32 via 4 WMMA, PV via 4 WMMA.
// Q and K tiles staged with async global->LDS; V transposed via LDS scatter.
// ---------------------------------------------------------------------------
__global__ __launch_bounds__(128)
void flash_attn_kernel(const bf16* __restrict__ Q, const bf16* __restrict__ Km,
                       const bf16* __restrict__ Vm, bf16* __restrict__ Oout) {
    constexpr int T = 2048, DK = 64;
    const int qb = blockIdx.x;          // 0..31
    const int bh = blockIdx.y;          // 0..31
    const int b  = bh >> 4;
    const int h  = bh & 15;

    const int tid  = threadIdx.x;
    const int lane = tid & 31;
    const int w    = tid >> 5;
    const int hl   = lane >> 4;
    const int ln   = lane & 15;

    __shared__ bf16 Qs[64 * 64];        // [q][dk]
    __shared__ bf16 Ks[32 * 64];        // [s][dk]
    __shared__ bf16 Vst[64 * 32];       // [dk][s]  (transposed for B-fragments)
    __shared__ bf16 Pw[4][16 * 32];     // per-wave P scratch [qrow][s]

    const bf16* Qg = Q  + (size_t)bh * T * DK;
    const bf16* Kg = Km + (size_t)bh * T * DK;
    const bf16* Vg = Vm + (size_t)bh * T * DK;

    const unsigned ldsQ0 = lds_off(Qs);
    const unsigned ldsK0 = lds_off(Ks);

    // Async-load the 64-row Q tile (contiguous 8 KB).
#pragma unroll
    for (int i = tid; i < 512; i += 128)
        async_ld16(ldsQ0 + (unsigned)i * 16u,
                   (unsigned)((size_t)qb * 64 * DK * sizeof(bf16)) + (unsigned)i * 16u,
                   (const void*)Qg);

    v8f o[4];
#pragma unroll
    for (int nt = 0; nt < 4; ++nt) o[nt] = zero_v8f();
    float mx[8], ls[8];
#pragma unroll
    for (int v = 0; v < 8; ++v) { mx[v] = -1e30f; ls[v] = 0.0f; }

    const float scale = 0.125f;              // 1/sqrt(64)
    const int qwave0 = qb * 64 + w * 16;
    const int nkb = 2 * (qb + 1);

    wait_async0();
    __syncthreads();

    for (int kb = 0; kb < nkb; ++kb) {
        const int s0 = kb * 32;
        // K block [32][64]: async straight copy.
#pragma unroll
        for (int i = tid; i < 256; i += 128)
            async_ld16(ldsK0 + (unsigned)i * 16u,
                       (unsigned)((size_t)s0 * DK * sizeof(bf16)) + (unsigned)i * 16u,
                       (const void*)Kg);
        // V block transposed into Vst[dk][s] (element scatter via VGPRs).
#pragma unroll
        for (int i = tid; i < 256; i += 128) {
            int s = i >> 3, c = (i & 7) * 8;
            uint4 d = *(const uint4*)(Vg + (size_t)(s0 + s) * DK + c);
            const bf16* e = (const bf16*)&d;
#pragma unroll
            for (int j = 0; j < 8; ++j) Vst[(c + j) * 32 + s] = e[j];
        }
        wait_async0();
        __syncthreads();

        if (s0 <= qwave0 + 15) {   // wave-uniform: any unmasked key in this block
            // ---- scores: two 16x16 tiles over 32 keys, K-dim = DK (2 chunks)
            v16bf aq[2];
#pragma unroll
            for (int kc2 = 0; kc2 < 2; ++kc2) {
                const bf16* ar = &Qs[(w * 16 + ln) * 64 + kc2 * 32 + hl * 8];
                aq[kc2] = load_frag2(ar, ar + 16);
            }
            v8f sc[2];
            sc[0] = zero_v8f(); sc[1] = zero_v8f();
#pragma unroll
            for (int st = 0; st < 2; ++st)
#pragma unroll
                for (int kc2 = 0; kc2 < 2; ++kc2) {
                    const bf16* br = &Ks[(st * 16 + ln) * 64 + kc2 * 32 + hl * 8];
                    v16bf bk = load_frag2(br, br + 16);
                    sc[st] = wmma_bf16(aq[kc2], bk, sc[st]);
                }

            // ---- online softmax (rows live across the 16 lanes of a half)
            float alpha[8];
#pragma unroll
            for (int v = 0; v < 8; ++v) {
                const int qg = qwave0 + 8 * hl + v;
                float x0 = sc[0][v] * scale;
                float x1 = sc[1][v] * scale;
                if (s0 + ln > qg)      x0 = -1e30f;
                if (s0 + 16 + ln > qg) x1 = -1e30f;
                float rm = fmaxf(x0, x1);
#pragma unroll
                for (int d = 1; d < 16; d <<= 1) rm = fmaxf(rm, __shfl_xor(rm, d, 32));
                const float mnew = fmaxf(mx[v], rm);
                const float a    = __expf(mx[v] - mnew);
                mx[v] = mnew;
                const float p0 = __expf(x0 - mnew);
                const float p1 = __expf(x1 - mnew);
                float rs = p0 + p1;
#pragma unroll
                for (int d = 1; d < 16; d <<= 1) rs += __shfl_xor(rs, d, 32);
                ls[v] = ls[v] * a + rs;
                alpha[v] = a;
                // C-layout -> LDS (row-major [qrow][s])
                Pw[w][(8 * hl + v) * 32 + ln]      = (bf16)p0;
                Pw[w][(8 * hl + v) * 32 + 16 + ln] = (bf16)p1;
            }
#pragma unroll
            for (int nt = 0; nt < 4; ++nt)
#pragma unroll
                for (int v = 0; v < 8; ++v) o[nt][v] *= alpha[v];

            // Same-wave LDS RAW: wait for the P stores before the A-layout reads.
            asm volatile("s_wait_dscnt 0x0" ::: "memory");

            const bf16* pr = &Pw[w][ln * 32 + hl * 8];
            v16bf pf = load_frag2(pr, pr + 16);
#pragma unroll
            for (int nt = 0; nt < 4; ++nt) {
                const bf16* br = &Vst[(nt * 16 + ln) * 32 + hl * 8];
                v16bf bv = load_frag2(br, br + 16);
                o[nt] = wmma_bf16(pf, bv, o[nt]);
            }
        }
        __syncthreads();
    }

    // ---- finalize: O /= l, scatter to attn[B*T][D] with heads interleaved
#pragma unroll
    for (int nt = 0; nt < 4; ++nt)
#pragma unroll
        for (int v = 0; v < 8; ++v) {
            const int qg = qwave0 + 8 * hl + v;
            const float val = o[nt][v] / ls[v];
            Oout[(size_t)(b * T + qg) * 1024 + h * 64 + nt * 16 + ln] = (bf16)val;
        }
}

// ---------------------------------------------------------------------------
extern "C" void kernel_launch(void* const* d_in, const int* in_sizes, int n_in,
                              void* d_out, int out_size, void* d_ws, size_t ws_size,
                              hipStream_t stream) {
    const float* x    = (const float*)d_in[0];
    const float* Wqkv = (const float*)d_in[1];
    const float* bqkv = (const float*)d_in[2];
    const float* Wout = (const float*)d_in[3];
    const float* bout = (const float*)d_in[4];
    float* out = (float*)d_out;

    const int B = 2, T = 2048, D = 1024, H = 16;
    const int M  = B * T;      // 4096
    const int N1 = 3 * D;      // 3072

    char* ws = (char*)d_ws;
    bf16* xbf    = (bf16*)(ws);                        //  8 MB  [M][D]
    bf16* WqT    = (bf16*)(ws + (size_t)( 8 << 20));   //  6 MB  [3D][D]
    bf16* WoT    = (bf16*)(ws + (size_t)(14 << 20));   //  2 MB  [D][D]
    bf16* qkvbf  = (bf16*)(ws + (size_t)(16 << 20));   // 24 MB  [M][3D]
    bf16* Qb     = (bf16*)(ws + (size_t)(40 << 20));   //  8 MB  [B*H][T][DK]
    bf16* Kb     = (bf16*)(ws + (size_t)(48 << 20));   //  8 MB
    bf16* Vb     = (bf16*)(ws + (size_t)(56 << 20));   //  8 MB
    bf16* attnbf = qkvbf;  // qkv is dead after rope_split; reuse for attn output

    // 0) precision conversion / weight transpose
    cast_f32_bf16_kernel<<<(M * D / 4 + 255) / 256, 256, 0, stream>>>(x, xbf, M * D / 4);
    transpose_cast_kernel<<<(int)(((long long)D * N1 + 255) / 256), 256, 0, stream>>>(
        Wqkv, WqT, D, N1);
    transpose_cast_kernel<<<(D * D + 255) / 256, 256, 0, stream>>>(Wout, WoT, D, D);

    // 1) qkv = x @ W_qkv + b_qkv   (bf16 out)
    gemm_kernel<true><<<dim3(N1 / 128, M / 128), 256, 0, stream>>>(
        xbf, WqT, bqkv, (void*)qkvbf, M, N1, D);

    // 2) RoPE + head split
    rope_split_kernel<<<(B * H * T * 32) / 256, 256, 0, stream>>>(qkvbf, Qb, Kb, Vb);

    // 3) causal flash attention
    flash_attn_kernel<<<dim3(T / 64, B * H), 128, 0, stream>>>(Qb, Kb, Vb, attnbf);

    // 4) out = attn @ W_out + b_out   (f32 out)
    gemm_kernel<false><<<dim3(D / 128, M / 128), 256, 0, stream>>>(
        attnbf, WoT, bout, (void*)out, M, D, D);
}
